// SPDD_12378095747666
// MI455X (gfx1250) — compile-verified, matrix operand order checked by reading, not run on hardware
//
#include <hip/hip_runtime.h>

typedef __attribute__((ext_vector_type(16))) _Float16 v16h;
typedef __attribute__((ext_vector_type(8)))  float    v8f;

#define THREADS 256

__global__ void spdd_init_kernel(unsigned int* __restrict__ ws) {
    if (threadIdx.x < 8) ws[threadIdx.x] = 0u;
}

// Exact replacement for sum_{t=0..49} (p > float(t)*0.02f):
// tau is strictly monotone, so the count is a boundary index. The trunc(p*50)
// guess is within +-2 of the boundary (f32 rel. error ~6e-8), and the fixups
// use the *same* comparison as the sweep, so the result is bit-identical.
__device__ __forceinline__ int tau_count(float p) {
    int c = (int)(p * 50.0f);
    c = c < 0 ? 0 : (c > 50 ? 50 : c);
    c += (c < 50 && p > (float)c * 0.02f) ? 1 : 0;
    c += (c < 50 && p > (float)c * 0.02f) ? 1 : 0;
    c -= (c > 0 && !(p > (float)(c - 1) * 0.02f)) ? 1 : 0;
    c -= (c > 0 && !(p > (float)(c - 1) * 0.02f)) ? 1 : 0;
    return c;
}

// Streaming pass: b128 loads, packed u64 register accumulation (no dynamic
// local-array indexing -> no scratch), WMMA block reduction, exact u32 atomics.
// Grid sized so each thread sees <= 9 elements: partials <= 450 are exact in
// f16, and the packed fields (sum:10 bits, count:6 bits) cannot overflow.
__global__ __launch_bounds__(THREADS)
void spdd_partial_kernel(const float* __restrict__ pred,
                         const int*   __restrict__ group,
                         unsigned int* __restrict__ ws, int n) {
    const int tid  = threadIdx.x;
    const int gtid = blockIdx.x * THREADS + tid;
    const int T    = gridDim.x * THREADS;

    // 4 groups x 16 bits each: bits[9:0] = sum of counts, bits[15:10] = #elems.
    unsigned long long acc = 0ull;

    const int n4 = n >> 2;
    for (int i4 = gtid; i4 < n4; i4 += T) {
        const float4 p4 = reinterpret_cast<const float4*>(pred)[i4];
        const int4   g4 = reinterpret_cast<const int4*>(group)[i4];
        const float pv[4] = {p4.x, p4.y, p4.z, p4.w};
        const int   gv[4] = {g4.x, g4.y, g4.z, g4.w};
        #pragma unroll
        for (int e = 0; e < 4; ++e) {
            const int c = tau_count(pv[e]);
            acc += (unsigned long long)(1024u + (unsigned)c) << ((gv[e] & 3) << 4);
        }
    }
    for (int i = (n & ~3) + gtid; i < n; i += T) {   // tail (n % 4 != 0)
        const int c = tau_count(pred[i]);
        acc += (unsigned long long)(1024u + (unsigned)c) << ((group[i] & 3) << 4);
    }

    // Staging: partsT[j][tid]; j = 0..3 group sums, 4..7 group counts,
    // 8..15 zero rows so the WMMA A-fragment loads are branch-free.
    __shared__ __align__(32) _Float16 partsT[16 * THREADS];
    #pragma unroll
    for (int j = 0; j < 4; ++j) {
        const unsigned int v   = (unsigned int)((acc >> (j * 16)) & 0xFFFFull);
        const unsigned int s   = v & 0x3FFu;
        const unsigned int cnt = v >> 10;
        partsT[j * THREADS + tid]       = (_Float16)(float)s;
        partsT[(4 + j) * THREADS + tid] = (_Float16)(float)cnt;
        partsT[(8 + j) * THREADS + tid]  = (_Float16)0.0f;
        partsT[(12 + j) * THREADS + tid] = (_Float16)0.0f;
    }
    __syncthreads();

    // Wave 0: 8 chained V_WMMA_F32_16X16X32_F16 with B = all-ones -> row sums.
    // With B == 1 only the multiset of values per row matters; row m is exactly
    // lane m's 16 halves  U  lane (m+16)'s 16 halves, so feeding lane (m, hi)
    // the 16 contiguous partials partsT[m][chunk*32 + hi*16 ..] accumulates
    // C[m,*] = sum over all 256 threads of partial m.
    if (tid < 32) {
        const int m  = tid & 15;   // matrix row this lane feeds
        const int hi = tid >> 4;   // which 16-thread half of the chunk

        v16h ones;
        #pragma unroll
        for (int i = 0; i < 16; ++i) ones[i] = (_Float16)1.0f;

        v8f c;
        #pragma unroll
        for (int i = 0; i < 8; ++i) c[i] = 0.0f;

        #pragma unroll
        for (int chunk = 0; chunk < THREADS / 32; ++chunk) {
            v16h a;
            const _Float16* src = &partsT[m * THREADS + chunk * 32 + hi * 16];
            #pragma unroll
            for (int i = 0; i < 16; ++i) a[i] = src[i];
            c = __builtin_amdgcn_wmma_f32_16x16x32_f16(
                    /*neg_a=*/false, a, /*neg_b=*/false, ones,
                    /*c_mod=*/(short)0, c, /*reuse_a=*/false, /*reuse_b=*/false);
        }
        // C layout: lanes 0..15 hold rows 0..7 in VGPRs 0..7; exact integers.
        if (tid == 0) {
            #pragma unroll
            for (int j = 0; j < 8; ++j)
                atomicAdd(&ws[j], (unsigned int)(c[j] + 0.5f));
        }
    }
}

__global__ void spdd_final_kernel(const unsigned int* __restrict__ ws,
                                  float* __restrict__ out) {
    if (threadIdx.x != 0 || blockIdx.x != 0) return;
    float parity[4];
    #pragma unroll
    for (int g = 0; g < 4; ++g) {
        const float S = (float)ws[g];
        const float C = (float)ws[4 + g];
        parity[g] = (S / (C + 1e-10f)) * (1.0f / 50.0f);
    }
    const int ii[6] = {0, 0, 0, 1, 1, 2};
    const int jj[6] = {1, 2, 3, 2, 3, 3};
    float sum = 0.0f, mx = 0.0f;
    #pragma unroll
    for (int k = 0; k < 6; ++k) {
        const float d = fabsf(parity[ii[k]] - parity[jj[k]]);
        sum += d;
        mx = d > mx ? d : mx;
    }
    out[0] = sum * (1.0f / 6.0f);
    out[1] = mx;
}

extern "C" void kernel_launch(void* const* d_in, const int* in_sizes, int n_in,
                              void* d_out, int out_size, void* d_ws, size_t ws_size,
                              hipStream_t stream) {
    const float* pred  = (const float*)d_in[0];
    // d_in[1] (target) is mathematically irrelevant to the reference result.
    const int*   group = (const int*)d_in[2];
    const int    n     = in_sizes[0];

    unsigned int* ws  = (unsigned int*)d_ws;
    float*        out = (float*)d_out;

    int blocks = (n + 2047) / 2048;   // <= 8 elems/thread in the vector loop
    if (blocks < 1) blocks = 1;

    hipLaunchKernelGGL(spdd_init_kernel,    dim3(1),      dim3(32),      0, stream, ws);
    hipLaunchKernelGGL(spdd_partial_kernel, dim3(blocks), dim3(THREADS), 0, stream,
                       pred, group, ws, n);
    hipLaunchKernelGGL(spdd_final_kernel,   dim3(1),      dim3(32),      0, stream, ws, out);
}
